// PointNet2MSGSeg_45904610460073
// MI455X (gfx1250) — compile-verified, matrix operand order checked by reading, not run on hardware
//
#include <hip/hip_runtime.h>

typedef __attribute__((ext_vector_type(16))) _Float16 v16h;
typedef __attribute__((ext_vector_type(8)))  _Float16 v8h;
typedef __attribute__((ext_vector_type(4)))  _Float16 v4h;
typedef __attribute__((ext_vector_type(8)))  float    v8f;

#define BB 4   // batch size

__device__ __forceinline__ void argmax_pair(float& v, int& i, float ov, int oi) {
  if (ov > v || (ov == v && oi < i)) { v = ov; i = oi; }
}

// ---------------------------------------------------------------------------
// Farthest point sampling: one 1024-thread block per batch, serial over S
// samples.  Wave32 __shfl_xor butterfly argmax + one 32-partial LDS stage.
// Tie -> lowest index (matching jnp.argmax).
// ---------------------------------------------------------------------------
__global__ __launch_bounds__(1024) void fps_kernel(
    const float* __restrict__ xyz, int N, int S,
    int* __restrict__ outIdx, float* __restrict__ distWS) {
  int b = blockIdx.x;
  const float* X = xyz + (size_t)b * N * 3;
  float* dist = distWS + (size_t)b * N;
  __shared__ float sV[32];
  __shared__ int   sI[32];
  __shared__ int   sFar;
  __shared__ float cen[3];
  int t = threadIdx.x;
  int lane = t & 31, wv = t >> 5;
  for (int i = t; i < N; i += 1024) dist[i] = 1e10f;
  if (t == 0) sFar = 0;
  __syncthreads();
  for (int s = 0; s < S; ++s) {
    int far = sFar;
    if (t == 0) outIdx[b * S + s] = far;
    if (t < 3) cen[t] = X[far * 3 + t];
    __syncthreads();
    float cx = cen[0], cy = cen[1], cz = cen[2];
    float best = -1.0f; int bi = 0;
    for (int i = t; i < N; i += 1024) {
      float dx = X[i*3+0]-cx, dy = X[i*3+1]-cy, dz = X[i*3+2]-cz;
      float d  = dx*dx + dy*dy + dz*dz;
      float nd = fminf(dist[i], d);
      dist[i] = nd;
      if (nd > best) { best = nd; bi = i; }
    }
    // in-wave butterfly argmax (wave32)
    for (int m = 16; m; m >>= 1) {
      float ov = __shfl_xor(best, m, 32);
      int   oi = __shfl_xor(bi,   m, 32);
      argmax_pair(best, bi, ov, oi);
    }
    if (lane == 0) { sV[wv] = best; sI[wv] = bi; }
    __syncthreads();
    if (wv == 0) {
      float v2 = sV[lane];
      int   i2 = sI[lane];
      for (int m = 16; m; m >>= 1) {
        float ov = __shfl_xor(v2, m, 32);
        int   oi = __shfl_xor(i2, m, 32);
        argmax_pair(v2, i2, ov, oi);
      }
      if (lane == 0) sFar = i2;
    }
    __syncthreads();
  }
}

__global__ void gather_xyz_k(const float* __restrict__ xyz, const int* __restrict__ idx,
                             int N, int S, float* __restrict__ out) {
  int i = blockIdx.x * 256 + threadIdx.x;
  if (i >= BB * S) return;
  int b = i / S;
  int j = idx[i];
  const float* p = xyz + ((size_t)b * N + j) * 3;
  out[(size_t)i*3+0] = p[0];
  out[(size_t)i*3+1] = p[1];
  out[(size_t)i*3+2] = p[2];
}

// Ball query: first K indices (ascending) within radius; pad with first hit.
__global__ void ball_query_k(const float* __restrict__ xyz, const float* __restrict__ nxyz,
                             int N, int S, int K, float r2, int* __restrict__ out) {
  int i = blockIdx.x * 256 + threadIdx.x;
  if (i >= BB * S) return;
  int b = i / S;
  const float* q = nxyz + (size_t)i * 3;
  const float* X = xyz + (size_t)b * N * 3;
  int* o = out + (size_t)i * K;
  int cnt = 0, first = -1;
  float qx = q[0], qy = q[1], qz = q[2];
  for (int j = 0; j < N; ++j) {
    float dx = qx - X[j*3], dy = qy - X[j*3+1], dz = qz - X[j*3+2];
    if (dx*dx + dy*dy + dz*dz <= r2) {
      if (first < 0) first = j;
      o[cnt++] = j;
      if (cnt == K) break;
    }
  }
  if (first < 0) first = 0;
  while (cnt < K) o[cnt++] = first;
}

// Build grouped feature rows (f16, K-padded; padding pre-zeroed by memset).
__global__ void group_k(const float* __restrict__ xyz, const float* __restrict__ nxyz,
                        const float* __restrict__ feats, int C,
                        const int* __restrict__ idx,
                        int N, int S, int K, int Kpad, int M,
                        _Float16* __restrict__ A) {
  int r = blockIdx.x * 256 + threadIdx.x;
  if (r >= M) return;
  int s = (r / K) % S;
  int b = r / (K * S);
  int j = idx[r];
  const float* p = xyz  + ((size_t)b * N + j) * 3;
  const float* q = nxyz + ((size_t)b * S + s) * 3;
  _Float16* a = A + (size_t)r * Kpad;
  if (feats) {
    const float* f = feats + ((size_t)b * N + j) * C;
    for (int c = 0; c < C; ++c) a[c] = (_Float16)f[c];
    a[C+0] = (_Float16)(p[0]-q[0]);
    a[C+1] = (_Float16)(p[1]-q[1]);
    a[C+2] = (_Float16)(p[2]-q[2]);
  } else {
    a[0] = (_Float16)(p[0]-q[0]);
    a[1] = (_Float16)(p[1]-q[1]);
    a[2] = (_Float16)(p[2]-q[2]);
    a[3] = (_Float16)p[0];
    a[4] = (_Float16)p[1];
    a[5] = (_Float16)p[2];
  }
}

// 3-NN inverse-distance interpolation + concat -> f16 A rows for FP modules.
__global__ void fp_build_k(const float* __restrict__ xyz1, const float* __restrict__ xyz2,
                           const float* __restrict__ pts1, int C1,
                           const float* __restrict__ pts2, int C2,
                           int N1, int N2, int Kpad, _Float16* __restrict__ A) {
  int r = blockIdx.x * 256 + threadIdx.x;
  if (r >= BB * N1) return;
  int b = r / N1;
  const float* q  = xyz1 + (size_t)r * 3;
  const float* X2 = xyz2 + (size_t)b * N2 * 3;
  float d0 = 1e30f, d1 = 1e30f, d2 = 1e30f;
  int   i0 = 0, i1 = 0, i2 = 0;
  float qx = q[0], qy = q[1], qz = q[2];
  for (int j = 0; j < N2; ++j) {
    float dx = qx - X2[j*3], dy = qy - X2[j*3+1], dz = qz - X2[j*3+2];
    float d = dx*dx + dy*dy + dz*dz;
    if (d < d0)      { d2=d1; i2=i1; d1=d0; i1=i0; d0=d; i0=j; }
    else if (d < d1) { d2=d1; i2=i1; d1=d; i1=j; }
    else if (d < d2) { d2=d; i2=j; }
  }
  d0 = fmaxf(d0, 0.f); d1 = fmaxf(d1, 0.f); d2 = fmaxf(d2, 0.f);
  float w0 = 1.f/(d0+1e-8f), w1 = 1.f/(d1+1e-8f), w2 = 1.f/(d2+1e-8f);
  float ws = w0 + w1 + w2;
  w0 /= ws; w1 /= ws; w2 /= ws;
  _Float16* a = A + (size_t)r * Kpad;
  if (pts1) {
    const float* p1 = pts1 + (size_t)r * C1;
    for (int c = 0; c < C1; ++c) a[c] = (_Float16)p1[c];
  }
  const float* p2 = pts2 + (size_t)b * N2 * C2;
  for (int c = 0; c < C2; ++c) {
    float v = w0 * p2[(size_t)i0*C2 + c] + w1 * p2[(size_t)i1*C2 + c] + w2 * p2[(size_t)i2*C2 + c];
    a[C1 + c] = (_Float16)v;
  }
}

// Convert f32 weight (Cin x Cout, row-major) -> f16 zero-padded (Kpad x Cpad).
__global__ void convert_w_k(const float* __restrict__ W, int Cin, int Cout,
                            int Kpad, int Cpad, _Float16* __restrict__ out) {
  int i = blockIdx.x * 256 + threadIdx.x;
  if (i >= Kpad * Cpad) return;
  int r = i / Cpad, c = i % Cpad;
  float v = (r < Cin && c < Cout) ? W[(size_t)r * Cout + c] : 0.f;
  out[i] = (_Float16)v;
}

// ---------------------------------------------------------------------------
// Fused WMMA GEMM + BN + ReLU.
//   A: M x Kpad f16 row-major.  W: Kpad x Cpad f16 row-major.
//   256 threads = 8 waves; each wave owns a 16M x 32N tile (two WMMA
//   accumulators sharing one A fragment); block covers 128M x 32N.
//   The 32-column weight panel is staged transposed into LDS in K-chunks of
//   512 up-front, so the WMMA inner loop is barrier-free and straight-line.
//   A-side prefetch is unconditional: default-TH prefetches are speculative
//   (dropped on translation failure, no counters), so no guard is needed.
//   flags bit0: BN+ReLU epilogue.  bit1: bias-only epilogue.
// ---------------------------------------------------------------------------
#define KCHUNK 512

__global__ __launch_bounds__(256) void gemm_bn_relu(
    const _Float16* __restrict__ A, int Kpad,
    const _Float16* __restrict__ W, int Cpad,
    const float* __restrict__ bias, const float* __restrict__ gamma,
    const float* __restrict__ beta, const float* __restrict__ mean,
    const float* __restrict__ var,
    int Cout, int flags,
    _Float16* __restrict__ outH, int ldo,
    float* __restrict__ outF, int ldoF) {
  extern __shared__ _Float16 ldsW[];           // [32][KC+8] transposed panel
  const int KC    = (Kpad < KCHUNK) ? Kpad : KCHUNK;
  const int KCP   = KC + 8;                    // row stride (16B-aligned)
  const int tid   = threadIdx.x;
  const int lane  = tid & 31;
  const int laneMod  = lane & 15;
  const int laneHalf = lane >> 4;
  const int wave  = tid >> 5;
  const int mBase = (blockIdx.x * 8 + wave) * 16;
  const int n0    = blockIdx.y * 32;

  v8f acc0 = {}, acc1 = {};
  const _Float16* aRow = A + (size_t)(mBase + laneMod) * Kpad + laneHalf * 8;
  const _Float16* lb0  = ldsW + (size_t)laneMod        * KCP + laneHalf * 8;
  const _Float16* lb1  = ldsW + (size_t)(16 + laneMod) * KCP + laneHalf * 8;

  for (int kb = 0; kb < Kpad; kb += KC) {
    const int kcur = (Kpad - kb < KC) ? (Kpad - kb) : KC;
    __syncthreads();   // protect previous chunk's LDS reads
    // stage W[kb..kb+kcur][n0..n0+31] transposed: ldsW[n][k]
    for (int e = tid * 4; e < kcur * 32; e += 1024) {
      int kk = e >> 5;
      int nn = e & 31;
      v4h wv = *(const v4h*)(W + (size_t)(kb + kk) * Cpad + (n0 + nn));
      ldsW[(nn+0)*KCP + kk] = wv[0];
      ldsW[(nn+1)*KCP + kk] = wv[1];
      ldsW[(nn+2)*KCP + kk] = wv[2];
      ldsW[(nn+3)*KCP + kk] = wv[3];
    }
    __syncthreads();

    for (int k0 = 0; k0 < kcur; k0 += 32) {
      // speculative prefetch of next A slice (safe past end-of-panel)
      __builtin_prefetch((const void*)(aRow + kb + k0 + 32), 0, 1);
      // A fragment (ISA 16-bit 16x32 layout): lanes 0-15 K{0-7,16-23}, 16-31 K{8-15,24-31}
      v8h a0 = *(const v8h*)(aRow + kb + k0);
      v8h a1 = *(const v8h*)(aRow + kb + k0 + 16);
      v16h af = __builtin_shufflevector(a0, a1, 0,1,2,3,4,5,6,7,8,9,10,11,12,13,14,15);
      // B fragments from transposed LDS panel (contiguous per-lane reads)
      v8h b0 = *(const v8h*)(lb0 + k0);
      v8h b1 = *(const v8h*)(lb0 + k0 + 16);
      v16h bf = __builtin_shufflevector(b0, b1, 0,1,2,3,4,5,6,7,8,9,10,11,12,13,14,15);
      acc0 = __builtin_amdgcn_wmma_f32_16x16x32_f16(false, af, false, bf,
                                                    (short)0, acc0, false, false);
      v8h c0 = *(const v8h*)(lb1 + k0);
      v8h c1 = *(const v8h*)(lb1 + k0 + 16);
      v16h cf = __builtin_shufflevector(c0, c1, 0,1,2,3,4,5,6,7,8,9,10,11,12,13,14,15);
      acc1 = __builtin_amdgcn_wmma_f32_16x16x32_f16(false, af, false, cf,
                                                    (short)0, acc1, false, false);
    }
  }

#pragma unroll
  for (int g = 0; g < 2; ++g) {
    v8f acc = g ? acc1 : acc0;
    const int col = n0 + g * 16 + laneMod;
    const bool valid = col < Cout;
    float sc = 0.f, sh = 0.f;
    if (valid) {
      if (flags & 1) {
        sc = gamma[col] * rsqrtf(var[col] + 1e-5f);
        sh = (bias[col] - mean[col]) * sc + beta[col];
      } else if (flags & 2) {
        sc = 1.f; sh = bias[col];
      } else {
        sc = 1.f;
      }
    }
#pragma unroll
    for (int i = 0; i < 8; ++i) {
      const int row = mBase + laneHalf * 8 + i;  // C/D 16x16 f32 VGPR layout
      float y = acc[i] * sc + sh;
      if (flags & 1) y = fmaxf(y, 0.f);
      if (outH) outH[(size_t)row * ldo + col] = (_Float16)(valid ? y : 0.f);
      if (outF && valid) outF[(size_t)row * ldoF + col] = y;
    }
  }
}

// Max over nsample dimension -> f32 feature buffer at channel offset coff.
__global__ void maxpool_k(const _Float16* __restrict__ act, int ldo, int K, int Cout,
                          int tot, float* __restrict__ out, int Ctot, int coff) {
  int i = blockIdx.x * 256 + threadIdx.x;
  if (i >= tot) return;
  int c = i % Cout;
  int s = i / Cout;
  const _Float16* p = act + (size_t)s * K * ldo + c;
  float m = (float)p[0];
  for (int k = 1; k < K; ++k) m = fmaxf(m, (float)p[(size_t)k * ldo]);
  out[(size_t)s * Ctot + coff + c] = m;
}

// ---------------------------------------------------------------------------
// Host orchestration
// ---------------------------------------------------------------------------
struct P {
  const float *W, *b, *be, *g, *m, *v;
  int Cin, Cout, Kpad, Cpad;
  _Float16* W16;
};

extern "C" void kernel_launch(void* const* d_in, const int* in_sizes, int n_in,
                              void* d_out, int out_size, void* d_ws, size_t ws_size,
                              hipStream_t stream) {
  (void)in_sizes; (void)n_in; (void)out_size; (void)ws_size;
  const float* data = (const float*)d_in[0];

  int ii = 1;
  auto rdL = [&](int cin, int cout) {
    P p;
    p.W  = (const float*)d_in[ii+0];
    p.b  = (const float*)d_in[ii+1];
    p.be = (const float*)d_in[ii+2];
    p.g  = (const float*)d_in[ii+3];
    p.m  = (const float*)d_in[ii+4];
    p.v  = (const float*)d_in[ii+5];
    ii += 6;
    p.Cin = cin; p.Cout = cout;
    p.Kpad = (cin + 31) & ~31;
    p.Cpad = (cout + 31) & ~31;
    p.W16 = nullptr;
    return p;
  };
  // pytree order: fp0, fp1, fp2, fp3, head1, head2, sa1, sa2, sa3, sa4
  P fp0L[2]; fp0L[0] = rdL(1536,512); fp0L[1] = rdL(512,512);
  P fp1L[2]; fp1L[0] = rdL(768,512);  fp1L[1] = rdL(512,512);
  P fp2L[2]; fp2L[0] = rdL(608,256);  fp2L[1] = rdL(256,256);
  P fp3L[2]; fp3L[0] = rdL(256,128);  fp3L[1] = rdL(128,128);
  P head1 = rdL(128,128);
  P head2;
  head2.W = (const float*)d_in[ii+0];
  head2.b = (const float*)d_in[ii+1];
  ii += 2;
  head2.be = head2.g = head2.m = head2.v = nullptr;
  head2.Cin = 128; head2.Cout = 13; head2.Kpad = 128; head2.Cpad = 32; head2.W16 = nullptr;
  P sa1b0[3]; sa1b0[0]=rdL(6,16);    sa1b0[1]=rdL(16,16);   sa1b0[2]=rdL(16,32);
  P sa1b1[3]; sa1b1[0]=rdL(6,32);    sa1b1[1]=rdL(32,32);   sa1b1[2]=rdL(32,64);
  P sa2b0[3]; sa2b0[0]=rdL(99,64);   sa2b0[1]=rdL(64,64);   sa2b0[2]=rdL(64,128);
  P sa2b1[3]; sa2b1[0]=rdL(99,64);   sa2b1[1]=rdL(64,96);   sa2b1[2]=rdL(96,128);
  P sa3b0[3]; sa3b0[0]=rdL(259,128); sa3b0[1]=rdL(128,196); sa3b0[2]=rdL(196,256);
  P sa3b1[3]; sa3b1[0]=rdL(259,128); sa3b1[1]=rdL(128,196); sa3b1[2]=rdL(196,256);
  P sa4b0[3]; sa4b0[0]=rdL(515,256); sa4b0[1]=rdL(256,256); sa4b0[2]=rdL(256,512);
  P sa4b1[3]; sa4b1[0]=rdL(515,256); sa4b1[1]=rdL(256,384); sa4b1[2]=rdL(384,512);

  // ---- workspace carve-up ----
  char* base = (char*)d_ws;
  size_t off = 0;
  auto alloc = [&](size_t bytes) -> void* {
    void* p = (void*)(base + off);
    off += (bytes + 255) & ~(size_t)255;
    return p;
  };
  float* distWS = (float*)alloc((size_t)BB * 16384 * 4);
  int*   fpsIdx = (int*)  alloc((size_t)BB * 1024 * 4);
  int*   ballIdx= (int*)  alloc((size_t)BB * 1024 * 32 * 4);
  float* l1x = (float*)alloc((size_t)BB * 1024 * 3 * 4);
  float* l2x = (float*)alloc((size_t)BB * 256  * 3 * 4);
  float* l3x = (float*)alloc((size_t)BB * 64   * 3 * 4);
  float* l4x = (float*)alloc((size_t)BB * 16   * 3 * 4);
  float* l1f = (float*)alloc((size_t)BB * 1024 * 96   * 4);
  float* l2f = (float*)alloc((size_t)BB * 256  * 256  * 4);
  float* l3f = (float*)alloc((size_t)BB * 64   * 512  * 4);
  float* l4f = (float*)alloc((size_t)BB * 16   * 1024 * 4);
  float* nl3 = (float*)alloc((size_t)BB * 64   * 512  * 4);
  float* nl2 = (float*)alloc((size_t)BB * 256  * 512  * 4);
  float* nl1 = (float*)alloc((size_t)BB * 1024 * 256  * 4);
  _Float16* actA = (_Float16*)alloc((size_t)65536  * 256 * 2);  // 33.5 MB
  _Float16* actB = (_Float16*)alloc((size_t)131072 * 64  * 2);  // 16.8 MB

  // ---- convert all weights to padded f16 ----
  P* allL[] = { &fp0L[0], &fp0L[1], &fp1L[0], &fp1L[1], &fp2L[0], &fp2L[1],
                &fp3L[0], &fp3L[1], &head1, &head2,
                &sa1b0[0], &sa1b0[1], &sa1b0[2], &sa1b1[0], &sa1b1[1], &sa1b1[2],
                &sa2b0[0], &sa2b0[1], &sa2b0[2], &sa2b1[0], &sa2b1[1], &sa2b1[2],
                &sa3b0[0], &sa3b0[1], &sa3b0[2], &sa3b1[0], &sa3b1[1], &sa3b1[2],
                &sa4b0[0], &sa4b0[1], &sa4b0[2], &sa4b1[0], &sa4b1[1], &sa4b1[2] };
  for (P* p : allL) {
    size_t n = (size_t)p->Kpad * p->Cpad;
    p->W16 = (_Float16*)alloc(n * 2);
    convert_w_k<<<(unsigned)((n + 255) / 256), 256, 0, stream>>>(
        p->W, p->Cin, p->Cout, p->Kpad, p->Cpad, p->W16);
  }

  auto runGemm = [&](const P& p, const _Float16* A, int M, int flags,
                     _Float16* oh, float* of, int ldoF) {
    dim3 grid(M / 128, p.Cpad / 32);
    int KC = (p.Kpad < KCHUNK) ? p.Kpad : KCHUNK;
    size_t shmem = (size_t)32 * (KC + 8) * 2;
    gemm_bn_relu<<<grid, 256, shmem, stream>>>(A, p.Kpad, p.W16, p.Cpad,
        p.b, p.g, p.be, p.m, p.v, p.Cout, flags, oh, p.Cpad, of, ldoF);
  };

  auto runSA = [&](const float* sxyz, int N, int S, const float* feats, int C,
                   float r0, int K0, P* br0, float r1, int K1, P* br1,
                   float* oxyz, float* ofeat, int Ctot) {
    fps_kernel<<<BB, 1024, 0, stream>>>(sxyz, N, S, fpsIdx, distWS);
    int q = BB * S;
    gather_xyz_k<<<(q + 255) / 256, 256, 0, stream>>>(sxyz, fpsIdx, N, S, oxyz);
    int coff = 0;
    for (int br = 0; br < 2; ++br) {
      P* L    = br ? br1 : br0;
      int K   = br ? K1 : K0;
      float r = br ? r1 : r0;
      ball_query_k<<<(q + 255) / 256, 256, 0, stream>>>(sxyz, oxyz, N, S, K, r * r, ballIdx);
      int M = q * K;
      hipMemsetAsync(actA, 0, (size_t)M * L[0].Kpad * 2, stream);
      group_k<<<(M + 255) / 256, 256, 0, stream>>>(sxyz, oxyz, feats, C, ballIdx,
                                                   N, S, K, L[0].Kpad, M, actA);
      runGemm(L[0], actA, M, 1, actB, nullptr, 0);
      runGemm(L[1], actB, M, 1, actA, nullptr, 0);
      runGemm(L[2], actA, M, 1, actB, nullptr, 0);
      int tot = q * L[2].Cout;
      maxpool_k<<<(tot + 255) / 256, 256, 0, stream>>>(actB, L[2].Cpad, K, L[2].Cout,
                                                       tot, ofeat, Ctot, coff);
      coff += L[2].Cout;
    }
  };

  auto runFP = [&](const float* x1, int N1, const float* x2, int N2,
                   const float* p1, int C1, const float* p2, int C2,
                   P* L, _Float16* finalH, float* finalF, int ldoF) {
    int M = BB * N1;
    hipMemsetAsync(actA, 0, (size_t)M * L[0].Kpad * 2, stream);
    fp_build_k<<<(M + 255) / 256, 256, 0, stream>>>(x1, x2, p1, C1, p2, C2,
                                                    N1, N2, L[0].Kpad, actA);
    runGemm(L[0], actA, M, 1, actB, nullptr, 0);
    runGemm(L[1], actB, M, 1, finalH, finalF, ldoF);
  };

  // ---- pipeline ----
  runSA(data, 16384, 1024, nullptr, 0,   0.05f, 16, sa1b0, 0.1f, 32, sa1b1, l1x, l1f, 96);
  runSA(l1x,  1024,  256,  l1f,   96,    0.1f,  16, sa2b0, 0.2f, 32, sa2b1, l2x, l2f, 256);
  runSA(l2x,  256,   64,   l2f,   256,   0.2f,  16, sa3b0, 0.4f, 32, sa3b1, l3x, l3f, 512);
  runSA(l3x,  64,    16,   l3f,   512,   0.4f,  16, sa4b0, 0.8f, 32, sa4b1, l4x, l4f, 1024);

  runFP(l3x, 64,    l4x, 16,   l3f,    512, l4f, 1024, fp0L, nullptr, nl3, 512);
  runFP(l2x, 256,   l3x, 64,   l2f,    256, nl3, 512,  fp1L, nullptr, nl2, 512);
  runFP(l1x, 1024,  l2x, 256,  l1f,    96,  nl2, 512,  fp2L, nullptr, nl1, 256);
  runFP(data, 16384, l1x, 1024, nullptr, 0,  nl1, 256,  fp3L, actA,    nullptr, 0);

  float* outX    = (float*)d_out;                       // (4,16384,13)
  float* outFeat = outX + (size_t)BB * 16384 * 13;      // (4,16384,128)
  runGemm(head1, actA, BB * 16384, 1, actB, outFeat, 128);
  runGemm(head2, actB, BB * 16384, 2, nullptr, outX, 13);
}
